// DecoderBlock_48799418417708
// MI455X (gfx1250) — matrix operand order repaired and checked by reading, not compiled, not run on hardware
//
#include <hip/hip_runtime.h>
#include <math.h>
#include <stdint.h>

typedef __attribute__((ext_vector_type(16))) __bf16       v16bf;
typedef __attribute__((ext_vector_type(8)))  float        v8f;
typedef __attribute__((ext_vector_type(4)))  unsigned int u32x4;

union AFrag { v16bf v; u32x4 q[2]; };

#define N_EMBD 2048
#define SEQ    2048
#define NH     16
#define HD     128
#define ROWS   4096   // B*T

// ---- CDNA5 async global->LDS copy (ASYNCcnt path), portable via inline asm ----
__device__ __forceinline__ unsigned lds_off(const void* p) {
  return (unsigned)(uintptr_t)p;   // generic LDS pointer: addr[31:0] = LDS byte offset
}
__device__ __forceinline__ void async_copy16(unsigned lds_byte, const void* g) {
  unsigned long long ga = (unsigned long long)(uintptr_t)g;
  asm volatile("global_load_async_to_lds_b128 %0, %1, off"
               :: "v"(lds_byte), "v"(ga) : "memory");
}
#define ASYNC_WAIT0() asm volatile("s_wait_asynccnt 0x0" ::: "memory")

__device__ __forceinline__ float half_reduce_max(float v) {
#pragma unroll
  for (int m = 1; m < 16; m <<= 1) v = fmaxf(v, __shfl_xor(v, m, 32));
  return v;
}
__device__ __forceinline__ float half_reduce_sum(float v) {
#pragma unroll
  for (int m = 1; m < 16; m <<= 1) v += __shfl_xor(v, m, 32);
  return v;
}
__device__ __forceinline__ float gelu_tanh(float x) {
  return 0.5f * x * (1.0f + tanhf(0.7978845608028654f * (x + 0.044715f * x * x * x)));
}

// ---------------- weight fp32 [K][N] -> bf16 transposed [N][K] ----------------
__global__ __launch_bounds__(256) void k_transpose_bf16(
    const float* __restrict__ W, __bf16* __restrict__ Wt, int K, int N) {
  __shared__ float tile[32][33];
  const int tx = threadIdx.x & 31, ty = threadIdx.x >> 5;   // 32x8
  const int n = blockIdx.x * 32 + tx;
  const int kbase = blockIdx.y * 32;
#pragma unroll
  for (int i = 0; i < 32; i += 8)
    tile[ty + i][tx] = W[(size_t)(kbase + ty + i) * N + n];
  __syncthreads();
  const int k = kbase + tx;
#pragma unroll
  for (int i = 0; i < 32; i += 8)
    Wt[(size_t)(blockIdx.x * 32 + ty + i) * K + k] = (__bf16)tile[tx][ty + i];
}

// ---------------- layernorm fp32 row -> bf16 ----------------
__global__ __launch_bounds__(256) void k_layernorm(
    const float* __restrict__ x, const float* __restrict__ g,
    const float* __restrict__ b, __bf16* __restrict__ out) {
  const int row = blockIdx.x;
  const int tid = threadIdx.x;
  const float* xr = x + (size_t)row * N_EMBD;
  float v[8]; float s = 0.f;
#pragma unroll
  for (int i = 0; i < 8; i++) { v[i] = xr[tid + i * 256]; s += v[i]; }
  __shared__ float red[256];
  red[tid] = s; __syncthreads();
  for (int off = 128; off > 0; off >>= 1) { if (tid < off) red[tid] += red[tid + off]; __syncthreads(); }
  const float mean = red[0] * (1.0f / N_EMBD);
  __syncthreads();
  float q = 0.f;
#pragma unroll
  for (int i = 0; i < 8; i++) { float d = v[i] - mean; q += d * d; }
  red[tid] = q; __syncthreads();
  for (int off = 128; off > 0; off >>= 1) { if (tid < off) red[tid] += red[tid + off]; __syncthreads(); }
  const float rstd = rsqrtf(red[0] * (1.0f / N_EMBD) + 1e-5f);
#pragma unroll
  for (int i = 0; i < 8; i++) {
    const int c = tid + i * 256;
    out[(size_t)row * N_EMBD + c] = (__bf16)((v[i] - mean) * rstd * g[c] + b[c]);
  }
}

// ---------------- generic bf16 WMMA GEMM (async double-buffered LDS) ----------------
// out[M,N] = epilogue( A[M,K](bf16) @ Bt[N,K]^T (bf16) + bias[N] )
// flags: bit0 = tanh-GELU, bit1 = bf16 output (else f32). residual may be null.
__global__ __launch_bounds__(256) void k_gemm(
    const __bf16* __restrict__ A, const __bf16* __restrict__ Bt,
    const float* __restrict__ bias, const float* __restrict__ residual,
    void* __restrict__ out, int M, int N, int K, int flags) {
  __shared__ __attribute__((aligned(16))) __bf16 As[2][128 * 32];
  __shared__ __attribute__((aligned(16))) __bf16 Bs[2][128 * 32];
  const int tid = threadIdx.x;
  const int wid = tid >> 5, lane = tid & 31, hh = lane >> 4, l = lane & 15;
  const int bm = blockIdx.y * 128, bn = blockIdx.x * 128;
  const int wm = (wid >> 2) * 64, wn = (wid & 3) * 32;
  v8f zero8 = {0.f, 0.f, 0.f, 0.f, 0.f, 0.f, 0.f, 0.f};
  v8f acc[4][2];
#pragma unroll
  for (int i = 0; i < 4; i++)
#pragma unroll
    for (int j = 0; j < 2; j++) acc[i][j] = zero8;

  const int grow = tid >> 2;         // 0..63
  const int gcol = (tid & 3) * 8;    // 0,8,16,24 (bf16 elems)
  const __bf16* Ap  = A  + (size_t)(bm + grow) * K + gcol;
  const __bf16* Ap2 = Ap + (size_t)64 * K;
  const __bf16* Bp  = Bt + (size_t)(bn + grow) * K + gcol;
  const __bf16* Bp2 = Bp + (size_t)64 * K;
  unsigned lA[2], lA2[2], lB[2], lB2[2];
#pragma unroll
  for (int s = 0; s < 2; s++) {
    lA[s]  = lds_off(&As[s][grow * 32 + gcol]);
    lA2[s] = lds_off(&As[s][(grow + 64) * 32 + gcol]);
    lB[s]  = lds_off(&Bs[s][grow * 32 + gcol]);
    lB2[s] = lds_off(&Bs[s][(grow + 64) * 32 + gcol]);
  }

  // prologue: tile 0 -> buffer 0
  async_copy16(lA[0],  Ap);
  async_copy16(lA2[0], Ap2);
  async_copy16(lB[0],  Bp);
  async_copy16(lB2[0], Bp2);

  const int nk = K >> 5;
  for (int ki = 0; ki < nk; ki++) {
    const int bufi = ki & 1;
    ASYNC_WAIT0();          // current tile resident in LDS
    __syncthreads();        // visible to all waves; all waves done reading buf^1
    if (ki + 1 < nk) {      // stream next tile into the other buffer
      const int k0 = (ki + 1) << 5;
      async_copy16(lA[bufi ^ 1],  Ap  + k0);
      async_copy16(lA2[bufi ^ 1], Ap2 + k0);
      async_copy16(lB[bufi ^ 1],  Bp  + k0);
      async_copy16(lB2[bufi ^ 1], Bp2 + k0);
    }
    AFrag af[4], bfr[2];
#pragma unroll
    for (int ms = 0; ms < 4; ms++) {
      const __bf16* p = &As[bufi][(wm + ms * 16 + l) * 32 + hh * 8]; // K {0..7|8..15}
      af[ms].q[0] = *(const u32x4*)p;
      af[ms].q[1] = *(const u32x4*)(p + 16);                         // K {16..23|24..31}
    }
#pragma unroll
    for (int ns = 0; ns < 2; ns++) {
      const __bf16* p = &Bs[bufi][(wn + ns * 16 + l) * 32 + hh * 8]; // K {0..7|8..15}
      bfr[ns].q[0] = *(const u32x4*)p;
      bfr[ns].q[1] = *(const u32x4*)(p + 16);                        // K {16..23|24..31}
    }
#pragma unroll
    for (int ms = 0; ms < 4; ms++)
#pragma unroll
      for (int ns = 0; ns < 2; ns++)
        acc[ms][ns] = __builtin_amdgcn_wmma_f32_16x16x32_bf16(
            false, af[ms].v, false, bfr[ns].v, (short)0, acc[ms][ns], false, false);
  }

  const bool do_gelu  = (flags & 1) != 0;
  const bool out_bf16 = (flags & 2) != 0;
#pragma unroll
  for (int ms = 0; ms < 4; ms++)
#pragma unroll
    for (int ns = 0; ns < 2; ns++)
#pragma unroll
      for (int v = 0; v < 8; v++) {
        const int row = bm + wm + ms * 16 + hh * 8 + v;
        const int col = bn + wn + ns * 16 + l;
        float val = acc[ms][ns][v] + bias[col];
        if (do_gelu) val = gelu_tanh(val);
        const size_t oi = (size_t)row * N + col;
        if (residual) val += residual[oi];
        if (out_bf16) ((__bf16*)out)[oi] = (__bf16)val;
        else          ((float*)out)[oi]  = val;
      }
}

// ---------------- split qkv (bf16) into Q,K [bh][t][d] and Vt [bh][d][t] ----------------
__global__ __launch_bounds__(256) void k_split_qkv(
    const __bf16* __restrict__ qkv, __bf16* __restrict__ Q,
    __bf16* __restrict__ Ko, __bf16* __restrict__ Vt) {
  const size_t i = (size_t)blockIdx.x * 256 + threadIdx.x;  // over bh*t*d
  const int d = (int)(i & 127);
  const size_t r = i >> 7;
  const int t = (int)(r & 2047);
  const int bh = (int)(r >> 11);
  const int b = bh >> 4, h = bh & 15;
  const size_t src = ((size_t)(b * SEQ + t)) * (3 * N_EMBD) + (size_t)h * 384;
  Q[i]  = qkv[src + d];
  Ko[i] = qkv[src + 128 + d];
  Vt[((size_t)bh * HD + d) * SEQ + t] = qkv[src + 256 + d];
}

// ---------------- flash attention: one (b,h,q-tile of 128) per block ----------------
__global__ __launch_bounds__(256) void k_attn(
    const __bf16* __restrict__ Q, const __bf16* __restrict__ Kg,
    const __bf16* __restrict__ Vt, __bf16* __restrict__ Out) {
  __shared__ __attribute__((aligned(16))) __bf16 sQ[128 * 128];     // Q tile, then P buffer
  __shared__ __attribute__((aligned(16))) __bf16 sK[2][128 * 128];  // double-buffered K
  __shared__ __attribute__((aligned(16))) __bf16 sV[2][128 * 128];  // double-buffered V^T
  const int qt = blockIdx.x, bh = blockIdx.y;
  const int b = bh >> 4, h = bh & 15;
  const int tid = threadIdx.x, wid = tid >> 5, lane = tid & 31;
  const int hh = lane >> 4, l = lane & 15;
  const int r0 = tid >> 4, c0 = (tid & 15) * 8;

  const __bf16* Qb   = Q  + ((size_t)bh * SEQ + qt * 128) * HD;
  const __bf16* Kbh  = Kg + (size_t)bh * SEQ * HD;
  const __bf16* Vbh  = Vt + (size_t)bh * HD * SEQ;

  // async-stage Q tile
#pragma unroll
  for (int p = 0; p < 8; p++)
    async_copy16(lds_off(sQ + (r0 + p * 16) * HD + c0),
                 Qb + (size_t)(r0 + p * 16) * HD + c0);

  // prologue: K/V tile 0 -> buffer 0
#pragma unroll
  for (int p = 0; p < 8; p++) {
    async_copy16(lds_off(&sK[0][(r0 + p * 16) * HD + c0]),
                 Kbh + (size_t)(r0 + p * 16) * HD + c0);
    async_copy16(lds_off(&sV[0][(r0 + p * 16) * HD + c0]),
                 Vbh + (size_t)(r0 + p * 16) * SEQ + c0);
  }
  ASYNC_WAIT0();
  __syncthreads();

  AFrag qf[4];   // wave's 16 q-rows, K(head dim)=128 as 4 x 32
#pragma unroll
  for (int kk = 0; kk < 4; kk++) {
    const __bf16* p = sQ + (wid * 16 + l) * HD + kk * 32 + hh * 8;
    qf[kk].q[0] = *(const u32x4*)p;
    qf[kk].q[1] = *(const u32x4*)(p + 16);
  }
  __syncthreads();   // sQ now free -> P buffer

  v8f zero8 = {0.f, 0.f, 0.f, 0.f, 0.f, 0.f, 0.f, 0.f};
  v8f accO[8];
#pragma unroll
  for (int i = 0; i < 8; i++) accO[i] = zero8;
  float mrow[8], lrow[8];
#pragma unroll
  for (int v = 0; v < 8; v++) { mrow[v] = -3.0e38f; lrow[v] = 0.f; }
  const float scale = 0.08838834764831845f;   // 1/sqrt(128)

  for (int kt = 0; kt <= qt; kt++) {
    const int bufi = kt & 1;
    // stream next K/V tile into the other buffer (all waves past the barrier,
    // so nobody is still reading buf^1 from iteration kt-1)
    if (kt + 1 <= qt) {
      const __bf16* Kb = Kbh + (size_t)(kt + 1) * 128 * HD;
      const __bf16* Vb = Vbh + (size_t)(kt + 1) * 128;
#pragma unroll
      for (int p = 0; p < 8; p++) {
        async_copy16(lds_off(&sK[bufi ^ 1][(r0 + p * 16) * HD + c0]),
                     Kb + (size_t)(r0 + p * 16) * HD + c0);
        async_copy16(lds_off(&sV[bufi ^ 1][(r0 + p * 16) * HD + c0]),
                     Vb + (size_t)(r0 + p * 16) * SEQ + c0);
      }
    }

    // S = Q @ K^T : 8 key-subtiles x 4 k-steps
    v8f accS[8];
#pragma unroll
    for (int i = 0; i < 8; i++) accS[i] = zero8;
#pragma unroll
    for (int kk = 0; kk < 4; kk++)
#pragma unroll
      for (int ns = 0; ns < 8; ns++) {
        AFrag bfr;
        const __bf16* p = &sK[bufi][(ns * 16 + l) * HD + kk * 32 + hh * 8];
        bfr.q[0] = *(const u32x4*)p;
        bfr.q[1] = *(const u32x4*)(p + 16);
        accS[ns] = __builtin_amdgcn_wmma_f32_16x16x32_bf16(
            false, qf[kk].v, false, bfr.v, (short)0, accS[ns], false, false);
      }

    // scale + causal mask + row max (rows live per (hh, v), cols per 16-lane half)
    float rmax[8];
#pragma unroll
    for (int v = 0; v < 8; v++) rmax[v] = -3.0e38f;
    const bool diag = (kt == qt);
#pragma unroll
    for (int ns = 0; ns < 8; ns++)
#pragma unroll
      for (int v = 0; v < 8; v++) {
        float s = accS[ns][v] * scale;
        if (diag && (ns * 16 + l > wid * 16 + hh * 8 + v)) s = -3.0e38f;
        accS[ns][v] = s;
        rmax[v] = fmaxf(rmax[v], s);
      }
#pragma unroll
    for (int v = 0; v < 8; v++) rmax[v] = half_reduce_max(rmax[v]);

    float corr[8];
#pragma unroll
    for (int v = 0; v < 8; v++) {
      const float mn = fmaxf(mrow[v], rmax[v]);
      corr[v] = __expf(mrow[v] - mn);
      mrow[v] = mn;
    }
    float rsum[8];
#pragma unroll
    for (int v = 0; v < 8; v++) rsum[v] = 0.f;
#pragma unroll
    for (int ns = 0; ns < 8; ns++)
#pragma unroll
      for (int v = 0; v < 8; v++) {
        const float pe = __expf(accS[ns][v] - mrow[v]);
        accS[ns][v] = pe;
        rsum[v] += pe;
      }
#pragma unroll
    for (int v = 0; v < 8; v++) {
      rsum[v] = half_reduce_sum(rsum[v]);
      lrow[v] = lrow[v] * corr[v] + rsum[v];
    }
#pragma unroll
    for (int nd = 0; nd < 8; nd++)
#pragma unroll
      for (int v = 0; v < 8; v++) accO[nd][v] *= corr[v];

    // P: C-layout -> A-layout via wave-private LDS region (no barrier needed)
#pragma unroll
    for (int ns = 0; ns < 8; ns++)
#pragma unroll
      for (int v = 0; v < 8; v++)
        sQ[(wid * 16 + hh * 8 + v) * 128 + ns * 16 + l] = (__bf16)accS[ns][v];

    // O += P @ V : K dim = 128 keys, N = head dim 128
#pragma unroll
    for (int kk = 0; kk < 4; kk++) {
      AFrag pf;
      const __bf16* pp = sQ + (wid * 16 + l) * 128 + kk * 32 + hh * 8;
      pf.q[0] = *(const u32x4*)pp;
      pf.q[1] = *(const u32x4*)(pp + 16);
#pragma unroll
      for (int nd = 0; nd < 8; nd++) {
        AFrag bfr;
        const __bf16* p = &sV[bufi][(nd * 16 + l) * HD + kk * 32 + hh * 8];
        bfr.q[0] = *(const u32x4*)p;
        bfr.q[1] = *(const u32x4*)(p + 16);
        accO[nd] = __builtin_amdgcn_wmma_f32_16x16x32_bf16(
            false, pf.v, false, bfr.v, (short)0, accO[nd], false, false);
      }
    }

    if (kt + 1 <= qt) {
      ASYNC_WAIT0();     // next K/V tile resident
      __syncthreads();   // single barrier per key-tile
    }
  }

#pragma unroll
  for (int v = 0; v < 8; v++) lrow[v] = 1.0f / lrow[v];
  const int trow = qt * 128 + wid * 16 + hh * 8;
#pragma unroll
  for (int nd = 0; nd < 8; nd++)
#pragma unroll
    for (int v = 0; v < 8; v++) {
      const size_t oi =
          ((size_t)(b * SEQ + trow + v)) * N_EMBD + (size_t)h * HD + nd * 16 + l;
      Out[oi] = (__bf16)(accO[nd][v] * lrow[v]);
    }
}

// ---------------- launch ----------------
extern "C" void kernel_launch(void* const* d_in, const int* in_sizes, int n_in,
                              void* d_out, int out_size, void* d_ws, size_t ws_size,
                              hipStream_t stream) {
  const float* x     = (const float*)d_in[0];
  const float* Wqkv  = (const float*)d_in[1];
  const float* bqkv  = (const float*)d_in[2];
  const float* Wproj = (const float*)d_in[3];
  const float* bproj = (const float*)d_in[4];
  const float* Wf1   = (const float*)d_in[5];
  const float* bf1   = (const float*)d_in[6];
  const float* Wf2   = (const float*)d_in[7];
  const float* bf2   = (const float*)d_in[8];
  const float* g1    = (const float*)d_in[9];
  const float* b1    = (const float*)d_in[10];
  const float* g2    = (const float*)d_in[11];
  const float* b2    = (const float*)d_in[12];

  char* ws = (char*)d_ws;
  size_t off = 0;
  auto alloc = [&](size_t bytes) -> char* {
    char* p = ws + off;
    off = (off + bytes + 255) & ~(size_t)255;
    return p;
  };
  __bf16* Wqkv_t  = (__bf16*)alloc((size_t)6144 * 2048 * 2);
  __bf16* Wproj_t = (__bf16*)alloc((size_t)2048 * 2048 * 2);
  __bf16* Wf1_t   = (__bf16*)alloc((size_t)8192 * 2048 * 2);
  __bf16* Wf2_t   = (__bf16*)alloc((size_t)2048 * 8192 * 2);
  __bf16* xn      = (__bf16*)alloc((size_t)ROWS * 2048 * 2);
  __bf16* qkvb    = (__bf16*)alloc((size_t)ROWS * 6144 * 2);
  __bf16* Qb      = (__bf16*)alloc((size_t)32 * SEQ * HD * 2);
  __bf16* Kb      = (__bf16*)alloc((size_t)32 * SEQ * HD * 2);
  __bf16* Vtb     = (__bf16*)alloc((size_t)32 * SEQ * HD * 2);
  __bf16* attnb   = (__bf16*)alloc((size_t)ROWS * 2048 * 2);
  float*  x1      = (float*)alloc((size_t)ROWS * 2048 * 4);
  __bf16* hbuf    = (__bf16*)alloc((size_t)ROWS * 8192 * 2);
  (void)ws_size; (void)in_sizes; (void)n_in; (void)out_size;

  dim3 blk(256);
  // weight convert + transpose (bf16, [N][K])
  k_transpose_bf16<<<dim3(6144 / 32, 2048 / 32), blk, 0, stream>>>(Wqkv, Wqkv_t, 2048, 6144);
  k_transpose_bf16<<<dim3(2048 / 32, 2048 / 32), blk, 0, stream>>>(Wproj, Wproj_t, 2048, 2048);
  k_transpose_bf16<<<dim3(8192 / 32, 2048 / 32), blk, 0, stream>>>(Wf1, Wf1_t, 2048, 8192);
  k_transpose_bf16<<<dim3(2048 / 32, 8192 / 32), blk, 0, stream>>>(Wf2, Wf2_t, 8192, 2048);

  // x -> ln1 -> qkv -> attention -> proj (+x) -> x1
  k_layernorm<<<ROWS, blk, 0, stream>>>(x, g1, b1, xn);
  k_gemm<<<dim3(6144 / 128, ROWS / 128), blk, 0, stream>>>(
      xn, Wqkv_t, bqkv, nullptr, qkvb, ROWS, 6144, 2048, /*bf16 out*/ 2);
  k_split_qkv<<<(32 * SEQ * HD) / 256, blk, 0, stream>>>(qkvb, Qb, Kb, Vtb);
  k_attn<<<dim3(SEQ / 128, 32), blk, 0, stream>>>(Qb, Kb, Vtb, attnb);
  k_gemm<<<dim3(2048 / 128, ROWS / 128), blk, 0, stream>>>(
      attnb, Wproj_t, bproj, x, x1, ROWS, 2048, 2048, /*f32 out*/ 0);

  // x1 -> ln2 -> gelu(ffn1) -> ffn2 (+x1) -> out
  k_layernorm<<<ROWS, blk, 0, stream>>>(x1, g2, b2, xn);
  k_gemm<<<dim3(8192 / 128, ROWS / 128), blk, 0, stream>>>(
      xn, Wf1_t, bf1, nullptr, hbuf, ROWS, 8192, 2048, /*gelu + bf16*/ 3);
  k_gemm<<<dim3(2048 / 128, ROWS / 128), blk, 0, stream>>>(
      hbuf, Wf2_t, bf2, x1, d_out, ROWS, 2048, 8192, /*f32 out*/ 0);
}
